// AbrialeLayer_73083163509298
// MI455X (gfx1250) — matrix operand order repaired and checked by reading, not compiled, not run on hardware
//
#include <hip/hip_runtime.h>
#include <stdint.h>
#include <math.h>

// ---------------- problem constants (from reference) ----------------
#define BB 4
#define TT 1024
#define DD 1024
#define HH 16
#define HD 64
#define DE 64
#define NT 8
#define NR 64
#define NH 4
#define NA 2
#define MM (BB*TT)          // 4096 rows
#define INV_SQ 0.125f       // 1/sqrt(HD)
#define INV_LOG_NT 0.48089834696298783f  // 1/log(8)

typedef __bf16 bf16;
typedef __attribute__((ext_vector_type(16))) __bf16    v16bf;
typedef __attribute__((ext_vector_type(8)))  float     v8f;
typedef __attribute__((ext_vector_type(8)))  uint32_t  v8u;
typedef __attribute__((ext_vector_type(4)))  float     v4f;
typedef __attribute__((ext_vector_type(8)))  uint16_t  v8us;
typedef __attribute__((ext_vector_type(4)))  int       v4i;

// ---------------- gfx1250 async global->LDS path (guarded) ----------------
#if defined(__AMDGCN__) && __has_builtin(__builtin_amdgcn_global_load_async_to_lds_b128) && __has_builtin(__builtin_amdgcn_s_wait_asynccnt)
#define USE_ASYNC_LDS 1
#define AS1 __attribute__((address_space(1)))
#define AS3 __attribute__((address_space(3)))
static __device__ __forceinline__ void async_cp16(const void* gsrc, void* ldst) {
  // Builtin signature (from hipcc diagnostic): (v4i AS1*, v4i AS3*, imm offset, imm cpol).
  // Global addresses are numerically identical in AS1; generic LDS addr[31:0] is the DS offset.
  __builtin_amdgcn_global_load_async_to_lds_b128((AS1 v4i*)(uintptr_t)gsrc,
                                                 (AS3 v4i*)(uint32_t)(uintptr_t)ldst, 0, 0);
}
static __device__ __forceinline__ void async_wait0() { __builtin_amdgcn_s_wait_asynccnt(0); }
#else
#define USE_ASYNC_LDS 0
#endif

static __device__ __forceinline__ v8f wmma_bf16(v16bf a, v16bf b, v8f c) {
  // v_wmma_f32_16x16x32_bf16: (neg_a, A, neg_b, B, c_mod, C, reuse_a, reuse_b)
  return __builtin_amdgcn_wmma_f32_16x16x32_bf16(false, a, false, b, (short)0, c, false, false);
}

// A-matrix 16x32 bf16 fragment; row data K-contiguous at `p` (dword ptr).
// lane m VGPR j: K = 2j + (j>=4 ? 8 : 0) + 8*hi -> dword (j<4 ? j : j+4) + 4*hi
static __device__ __forceinline__ v16bf fragA(const uint32_t* p, int hi) {
  v8u t;
#pragma unroll
  for (int j = 0; j < 8; ++j) t[j] = p[((j < 4) ? j : (j + 4)) + 4 * hi];
  return __builtin_bit_cast(v16bf, t);
}
// B-matrix 32x16 bf16 fragment; column n's K-contiguous data at `p` (dword ptr).
// lane n VGPR j: K = 2j + (j>=4 ? 8 : 0) + 8*hi -> dword (j<4 ? j : j+4) + 4*hi
static __device__ __forceinline__ v16bf fragB(const uint32_t* p, int hi) {
  v8u t;
#pragma unroll
  for (int j = 0; j < 8; ++j) t[j] = p[((j < 4) ? j : (j + 4)) + 4 * hi];
  return __builtin_bit_cast(v16bf, t);
}

static __device__ __forceinline__ float fast_rcp(float x) {
#if defined(__AMDGCN__)
  return __builtin_amdgcn_rcpf(x);
#else
  return 1.f / x;
#endif
}
static __device__ __forceinline__ float sigmoidf_(float x) { return fast_rcp(1.f + __expf(-x)); }

// ---------------- generic WMMA GEMM with fused epilogues ----------------
struct GemmEpi {
  float* outF; bf16* outB;
  const float* addF;          // EPI_ADDX: residual add
  const float* prevF;         // EPI_GATE: previously accumulated partial
  const float* rowscale;      // EPI_RS*: per-row scale
  int rs_stride;
  const float* ent;           // EPI_GATE: per-row entropy_norm
  const float* wlast;         // EPI_GATE: Wg1 last row (length N)
  const float* bias;          // EPI_GATE: bg1
};
enum { EPI_BF16 = 0, EPI_ADDX = 1, EPI_F32 = 2, EPI_RS0 = 3, EPI_RS1 = 4, EPI_GATE = 5 };

template <int EPI, bool ABF16, int BM, int BN, int WM, int WN>
__launch_bounds__((BM / WM) * (BN / WN) * 32)
__global__ void gemm_wmma(const void* Aptr, const float* Bm, int N, int K, GemmEpi e) {
  constexpr int NWAVE = (BM / WM) * (BN / WN);
  constexpr int NTHR  = NWAVE * 32;
  constexpr int WCOLS = BN / WN;
  constexpr int FM = WM / 16, FN = WN / 16;
  __shared__ __align__(16) bf16 As[BM * 32];   // (m,k) row-major
  __shared__ __align__(16) bf16 Bs[BN * 32];   // (n,k) transposed
  const int tid = threadIdx.x, lane = tid & 31, wid = tid >> 5;
  const int lrow = lane & 15, hi = lane >> 4;
  const int m0 = blockIdx.y * BM, n0 = blockIdx.x * BN;
  const int wm = (wid / WCOLS) * WM, wn = (wid % WCOLS) * WN;
  v8f acc[FM][FN] = {};

  for (int k0 = 0; k0 < K; k0 += 32) {
    if (k0 + 32 < K) {   // prefetch next k-slab (global_prefetch_b8)
      if (ABF16) __builtin_prefetch((const char*)Aptr + ((size_t)(m0 + (tid & (BM - 1))) * K + k0 + 32) * 2, 0, 1);
      else       __builtin_prefetch((const float*)Aptr + (size_t)(m0 + (tid & (BM - 1))) * K + k0 + 32, 0, 1);
      __builtin_prefetch(Bm + (size_t)(k0 + 32 + (tid & 31)) * N + n0, 0, 1);
    }
    __syncthreads();
    if constexpr (ABF16) {
#if USE_ASYNC_LDS
#pragma unroll
      for (int i = tid; i < BM * 4; i += NTHR) {           // 16B chunks
        int r = i >> 2, off = (i & 3) * 16;
        async_cp16((const char*)Aptr + ((size_t)(m0 + r) * K + k0) * 2 + off,
                   (char*)As + r * 64 + off);
      }
#else
      const bf16* A = (const bf16*)Aptr;
#pragma unroll
      for (int i = tid; i < BM * 4; i += NTHR) {
        int r = i >> 2, c = (i & 3) * 8;
        *(v8us*)((uint16_t*)As + r * 32 + c) =
            *(const v8us*)((const uint16_t*)(A + (size_t)(m0 + r) * K + k0) + c);
      }
#endif
    } else {
      const float* A = (const float*)Aptr;
#pragma unroll
      for (int i = tid; i < BM * 8; i += NTHR) {           // 4-float chunks, convert
        int r = i >> 3, c = (i & 7) * 4;
        v4f v = *(const v4f*)(A + (size_t)(m0 + r) * K + k0 + c);
        bf16* d = As + r * 32 + c;
        d[0] = (bf16)v[0]; d[1] = (bf16)v[1]; d[2] = (bf16)v[2]; d[3] = (bf16)v[3];
      }
    }
    // B is always f32 weights: convert + transpose into (n,k)
#pragma unroll
    for (int i = tid; i < BN * 8; i += NTHR) {
      int kk = i / (BN / 4), nn = (i % (BN / 4)) * 4;
      v4f v = *(const v4f*)(Bm + (size_t)(k0 + kk) * N + n0 + nn);
      Bs[(nn + 0) * 32 + kk] = (bf16)v[0];
      Bs[(nn + 1) * 32 + kk] = (bf16)v[1];
      Bs[(nn + 2) * 32 + kk] = (bf16)v[2];
      Bs[(nn + 3) * 32 + kk] = (bf16)v[3];
    }
#if USE_ASYNC_LDS
    if constexpr (ABF16) async_wait0();
#endif
    __syncthreads();

    v16bf af[FM], bfr[FN];
#pragma unroll
    for (int fm = 0; fm < FM; ++fm)
      af[fm] = fragA((const uint32_t*)As + (wm + fm * 16 + lrow) * 16, hi);
#pragma unroll
    for (int fn = 0; fn < FN; ++fn)
      bfr[fn] = fragB((const uint32_t*)Bs + (wn + fn * 16 + lrow) * 16, hi);
#pragma unroll
    for (int fm = 0; fm < FM; ++fm)
#pragma unroll
      for (int fn = 0; fn < FN; ++fn)
        acc[fm][fn] = wmma_bf16(af[fm], bfr[fn], acc[fm][fn]);
  }

#pragma unroll
  for (int fm = 0; fm < FM; ++fm)
#pragma unroll
    for (int fn = 0; fn < FN; ++fn)
#pragma unroll
      for (int r = 0; r < 8; ++r) {
        int m = m0 + wm + fm * 16 + r + 8 * hi;
        int n = n0 + wn + fn * 16 + lrow;
        size_t off = (size_t)m * N + n;
        float v = acc[fm][fn][r];
        if constexpr (EPI == EPI_BF16) {
          e.outB[off] = (bf16)v;
        } else if constexpr (EPI == EPI_ADDX) {
          float s = v + e.addF[off];
          e.outF[off] = s; e.outB[off] = (bf16)s;
        } else if constexpr (EPI == EPI_F32) {
          e.outF[off] = v;
        } else if constexpr (EPI == EPI_RS0) {
          e.outF[off] = e.rowscale[(size_t)m * e.rs_stride] * v;
        } else if constexpr (EPI == EPI_RS1) {
          float s = e.outF[off] + e.rowscale[(size_t)m * e.rs_stride] * v;
          e.outF[off] = s; e.outB[off] = (bf16)s;
        } else {  // EPI_GATE: silu(acc + prev + ent[m]*wlast[n] + bias[n]) -> bf16
          float s = v + e.prevF[off] + e.ent[m] * e.wlast[n] + e.bias[n];
          e.outB[off] = (bf16)(s * sigmoidf_(s));
        }
      }
}

// ---------------- ax = einsum('bhtd,hd->bht') ----------------
__global__ void ax_kernel(const bf16* nodes, const float* arity, float* ax) {
  int idx = blockIdx.x * blockDim.x + threadIdx.x;   // b*16384 + h*1024 + t
  int t = idx & (TT - 1);
  int h = (idx >> 10) & (HH - 1);
  int b = idx >> 14;
  const bf16* p = nodes + (size_t)(b * TT + t) * DD + h * HD;
  float s = 0.f;
#pragma unroll 8
  for (int d = 0; d < HD; ++d) s += (float)p[d] * arity[h * HD + d];
  ax[idx] = s;
}

// ---------------- fused flash attention with ternary-axis modulation ----------------
__launch_bounds__(128)
__global__ void flash_attn(const bf16* nodes, const bf16* values, const float* ax, bf16* attn) {
  __shared__ __align__(16) bf16 Kb[32 * 64];      // key block (s, hd)
  __shared__ __align__(16) bf16 Vt[64 * 32];      // value block transposed (hd, s)
  __shared__ __align__(16) bf16 Pl[4][16 * 32];   // per-wave P tile (layout round-trip)
  const int tid = threadIdx.x, lane = tid & 31, wid = tid >> 5;
  const int lrow = lane & 15, hi = lane >> 4;
  const int bh = blockIdx.y, b = bh >> 4, h = bh & 15;
  const int q0 = blockIdx.x * 64 + wid * 16;
  const bf16* nb = nodes + (size_t)b * TT * DD + h * HD;
  const bf16* vb = values + (size_t)b * TT * DD + h * HD;
  const float* axp = ax + (size_t)bh * TT;

  v16bf qf0, qf1;
  {
    const uint32_t* qp = (const uint32_t*)(nb + (size_t)(q0 + lrow) * DD);
    qf0 = fragA(qp, hi);
    qf1 = fragA(qp + 16, hi);
  }
  float axq[8];
#pragma unroll
  for (int r = 0; r < 8; ++r) axq[r] = axp[q0 + r + 8 * hi] * INV_SQ;
  float rmax[8], rsum[8];
#pragma unroll
  for (int r = 0; r < 8; ++r) { rmax[r] = -1e30f; rsum[r] = 0.f; }
  v8f O[4] = {};

  for (int s0 = 0; s0 < TT; s0 += 32) {
    __syncthreads();
#if USE_ASYNC_LDS
#pragma unroll
    for (int i = tid; i < 256; i += 128) {                 // 32 rows * 128B, 16B chunks
      int r = i >> 3, off = (i & 7) * 16;
      async_cp16((const char*)(nb + (size_t)(s0 + r) * DD) + off, (char*)Kb + r * 128 + off);
    }
#else
#pragma unroll
    for (int i = tid; i < 256; i += 128) {
      int r = i >> 3, c = (i & 7) * 8;
      *(v8us*)((uint16_t*)Kb + r * 64 + c) =
          *(const v8us*)((const uint16_t*)(nb + (size_t)(s0 + r) * DD) + c);
    }
#endif
    // Vt needs a transpose: vector global load, scattered b16 LDS stores
#pragma unroll
    for (int i = tid; i < 256; i += 128) {
      int r = i >> 3, c = (i & 7) * 8;
      v8us v = *(const v8us*)((const uint16_t*)(vb + (size_t)(s0 + r) * DD) + c);
#pragma unroll
      for (int j = 0; j < 8; ++j) ((uint16_t*)Vt)[(c + j) * 32 + r] = v[j];
    }
#if USE_ASYNC_LDS
    async_wait0();
#endif
    __syncthreads();

    // Preload ALL four K fragments into distinct registers so the ds_loads
    // pipeline, then issue the four WMMAs back-to-back (no per-WMMA dscnt wait).
    v16bf bk00 = fragB((const uint32_t*)Kb + lrow * 32, hi);
    v16bf bk01 = fragB((const uint32_t*)Kb + lrow * 32 + 16, hi);
    v16bf bk10 = fragB((const uint32_t*)Kb + (16 + lrow) * 32, hi);
    v16bf bk11 = fragB((const uint32_t*)Kb + (16 + lrow) * 32 + 16, hi);
    v8f S0 = {}, S1 = {};
    S0 = wmma_bf16(qf0, bk00, S0);
    S0 = wmma_bf16(qf1, bk01, S0);
    S1 = wmma_bf16(qf0, bk10, S1);
    S1 = wmma_bf16(qf1, bk11, S1);

    float axs0 = axp[s0 + lrow] * INV_SQ;
    float axs1 = axp[s0 + 16 + lrow] * INV_SQ;
    bf16* pw = Pl[wid];
#pragma unroll
    for (int r = 0; r < 8; ++r) {
      float e0 = S0[r] * INV_SQ * sigmoidf_(axq[r] + axs0);
      float e1 = S1[r] * INV_SQ * sigmoidf_(axq[r] + axs1);
      float mx = fmaxf(e0, e1);
#pragma unroll
      for (int d = 1; d < 16; d <<= 1) mx = fmaxf(mx, __shfl_xor(mx, d, 32));
      float mnew = fmaxf(rmax[r], mx);
      float alpha = __expf(rmax[r] - mnew);
      rmax[r] = mnew;
      float p0 = __expf(e0 - mnew), p1 = __expf(e1 - mnew);
      float ps = p0 + p1;
#pragma unroll
      for (int d = 1; d < 16; d <<= 1) ps += __shfl_xor(ps, d, 32);
      rsum[r] = rsum[r] * alpha + ps;
#pragma unroll
      for (int c = 0; c < 4; ++c) O[c][r] *= alpha;
      pw[(r + 8 * hi) * 32 + lrow] = (bf16)p0;
      pw[(r + 8 * hi) * 32 + 16 + lrow] = (bf16)p1;
    }
    // Preload P fragment and all four V fragments, then 4 WMMAs back-to-back.
    v16bf pf  = fragA((const uint32_t*)pw + lrow * 16, hi);
    v16bf bv0 = fragB((const uint32_t*)Vt + (0  + lrow) * 16, hi);
    v16bf bv1 = fragB((const uint32_t*)Vt + (16 + lrow) * 16, hi);
    v16bf bv2 = fragB((const uint32_t*)Vt + (32 + lrow) * 16, hi);
    v16bf bv3 = fragB((const uint32_t*)Vt + (48 + lrow) * 16, hi);
    O[0] = wmma_bf16(pf, bv0, O[0]);
    O[1] = wmma_bf16(pf, bv1, O[1]);
    O[2] = wmma_bf16(pf, bv2, O[2]);
    O[3] = wmma_bf16(pf, bv3, O[3]);
  }

#pragma unroll
  for (int r = 0; r < 8; ++r) {
    float inv = fast_rcp(rsum[r]);
    int t = q0 + r + 8 * hi;
#pragma unroll
    for (int c = 0; c < 4; ++c)
      attn[(size_t)(b * TT + t) * DD + h * HD + 16 * c + lrow] = (bf16)(O[c][r] * inv);
  }
}

// ---------------- normalized patterns ----------------
__global__ void pn_prep(const float* patterns, float* pn) {
  int r = blockIdx.x * blockDim.x + threadIdx.x;
  if (r >= NR) return;
  float s = 0.f;
  for (int d = 0; d < DE; ++d) { float v = patterns[r * DE + d]; s += v * v; }
  float inv = fast_rcp(fmaxf(sqrtf(s), 1e-12f));
  for (int d = 0; d < DE; ++d) pn[r * DE + d] = patterns[r * DE + d] * inv;
}

// ---------------- rule bank + event-type entropy (one wave per token) ----------------
__launch_bounds__(128)
__global__ void rule_kernel(const float* events, const float* pn, const float* patterns,
                            const float* W_alt, const float* W_type, const bf16* x1b,
                            const float* log_temp, float* rs, float* ent) {
  __shared__ float enS[4][DE];
  const int tid = threadIdx.x, lane = tid & 31, wid = tid >> 5;
  const int m = blockIdx.x * 4 + wid;

  float e0 = events[(size_t)m * DE + lane];
  float e1 = events[(size_t)m * DE + 32 + lane];
  float ss = e0 * e0 + e1 * e1;
#pragma unroll
  for (int d = 1; d < 32; d <<= 1) ss += __shfl_xor(ss, d, 32);
  float inv = fast_rcp(fmaxf(sqrtf(ss), 1e-12f));
  enS[wid][lane] = e0 * inv;
  enS[wid][32 + lane] = e1 * inv;

  float sim0 = 0.f, sim1 = 0.f;
  for (int d = 0; d < DE; ++d) {
    float ev = enS[wid][d];
    sim0 += ev * pn[lane * DE + d];
    sim1 += ev * pn[(lane + 32) * DE + d];
  }

  float topv[NH]; int topi[NH];
#pragma unroll
  for (int k = 0; k < NH; ++k) {
    float v; int idx;
    if (sim0 >= sim1) { v = sim0; idx = lane; } else { v = sim1; idx = lane + 32; }
#pragma unroll
    for (int d = 1; d < 32; d <<= 1) {
      float ov = __shfl_xor(v, d, 32);
      int   oi = __shfl_xor(idx, d, 32);
      if (ov > v || (ov == v && oi < idx)) { v = ov; idx = oi; }
    }
    topv[k] = v; topi[k] = idx;
    if (idx == lane) sim0 = -1e30f;
    else if (idx == lane + 32) sim1 = -1e30f;
  }

  float temp = fminf(fmaxf(__expf(log_temp[0]), 0.01f), 10.f);
  float it = fast_rcp(temp);
  float hw[NH]; float hsum = 0.f;
#pragma unroll
  for (int k = 0; k < NH; ++k) { hw[k] = __expf((topv[k] - topv[0]) * it); hsum += hw[k]; }
  float rhs = fast_rcp(hsum);
#pragma unroll
  for (int k = 0; k < NH; ++k) hw[k] *= rhs;

  float wp0 = 0.f, wp1 = 0.f;
#pragma unroll
  for (int k = 0; k < NH; ++k) {
    const float* pp = patterns + (size_t)topi[k] * DE;
    wp0 += hw[k] * pp[lane];
    wp1 += hw[k] * pp[32 + lane];
  }
  float pa0 = wp0 * W_alt[lane * NA + 0] + wp1 * W_alt[(lane + 32) * NA + 0];
  float pa1 = wp0 * W_alt[lane * NA + 1] + wp1 * W_alt[(lane + 32) * NA + 1];
#pragma unroll
  for (int d = 1; d < 32; d <<= 1) { pa0 += __shfl_xor(pa0, d, 32); pa1 += __shfl_xor(pa1, d, 32); }
  float l0 = pa0 * it, l1 = pa1 * it;
  float lm = fmaxf(l0, l1);
  float a0 = __expf(l0 - lm), a1 = __expf(l1 - lm);
  float ras = fast_rcp(a0 + a1);
  float hsig = sigmoidf_(topv[0]);
  if (lane == 0) {
    rs[(size_t)m * NA + 0] = a0 * ras * hsig;
    rs[(size_t)m * NA + 1] = a1 * ras * hsig;
  }

  float acc[NT];
#pragma unroll
  for (int n = 0; n < NT; ++n) acc[n] = 0.f;
  const bf16* xr = x1b + (size_t)m * DD;
  for (int kk = 0; kk < 32; ++kk) {
    int k = lane + 32 * kk;
    float xv = (float)xr[k];
    const float* wt = W_type + (size_t)k * NT;
#pragma unroll
    for (int n = 0; n < NT; ++n) acc[n] += xv * wt[n];
  }
#pragma unroll
  for (int n = 0; n < NT; ++n)
#pragma unroll
    for (int d = 1; d < 32; d <<= 1) acc[n] += __shfl_xor(acc[n], d, 32);
  float lmx = acc[0];
#pragma unroll
  for (int n = 1; n < NT; ++n) lmx = fmaxf(lmx, acc[n]);
  float es = 0.f; float pex[NT];
#pragma unroll
  for (int n = 0; n < NT; ++n) { pex[n] = __expf(acc[n] - lmx); es += pex[n]; }
  float res = fast_rcp(es);
  float H = 0.f;
#pragma unroll
  for (int n = 0; n < NT; ++n) { float p = pex[n] * res; H -= p * __logf(p + 1e-10f); }
  if (lane == 0) ent[m] = H * INV_LOG_NT;
}

// ---------------- final: g = sigmoid(h@Wg2+bg2); out = x1 + g*actions ----------------
__launch_bounds__(256)
__global__ void final_kernel(const bf16* hbf, const float* Wg2, const float* bg2,
                             const float* x1, const float* actions, float* out) {
  const int tid = threadIdx.x, lane = tid & 31, wid = tid >> 5;
  const int m = blockIdx.x * 8 + wid;
  const bf16* hr = hbf + (size_t)m * DD;
  float p = 0.f;
  for (int kk = 0; kk < 32; ++kk) { int k = lane + 32 * kk; p += (float)hr[k] * Wg2[k]; }
#pragma unroll
  for (int d = 1; d < 32; d <<= 1) p += __shfl_xor(p, d, 32);
  float g = sigmoidf_(p + bg2[0]);
  size_t base = (size_t)m * DD;
  for (int kk = 0; kk < 32; ++kk) { int k = lane + 32 * kk; out[base + k] = x1[base + k] + g * actions[base + k]; }
}

// ---------------- host orchestration ----------------
extern "C" void kernel_launch(void* const* d_in, const int* in_sizes, int n_in,
                              void* d_out, int out_size, void* d_ws, size_t ws_size,
                              hipStream_t stream) {
  (void)in_sizes; (void)n_in; (void)out_size; (void)ws_size;
  const float* x        = (const float*)d_in[0];
  const float* W_node   = (const float*)d_in[1];
  const float* W_value  = (const float*)d_in[2];
  const float* W_out    = (const float*)d_in[3];
  const float* arity_w  = (const float*)d_in[4];
  const float* W_event  = (const float*)d_in[5];
  const float* W_type   = (const float*)d_in[6];
  const float* patterns = (const float*)d_in[7];
  const float* W_actions= (const float*)d_in[8];
  const float* W_alt    = (const float*)d_in[9];
  const float* log_temp = (const float*)d_in[10];
  const float* Wg1      = (const float*)d_in[11];
  const float* bg1      = (const float*)d_in[12];
  const float* Wg2      = (const float*)d_in[13];
  const float* bg2      = (const float*)d_in[14];
  float* out = (float*)d_out;

  char* wp = (char*)d_ws;
  auto alloc = [&wp](size_t bytes) -> char* {
    char* p = wp; wp += (bytes + 255) & ~(size_t)255; return p;
  };
  bf16*  nodes_b  = (bf16*)alloc((size_t)MM * DD * 2);
  bf16*  values_b = (bf16*)alloc((size_t)MM * DD * 2);
  float* axw      = (float*)alloc((size_t)BB * HH * TT * 4);
  bf16*  attn_b   = (bf16*)alloc((size_t)MM * DD * 2);
  float* x1f      = (float*)alloc((size_t)MM * DD * 4);
  bf16*  x1b      = (bf16*)alloc((size_t)MM * DD * 2);
  float* events   = (float*)alloc((size_t)MM * DE * 4);
  float* pnw      = (float*)alloc((size_t)NR * DE * 4);
  float* rsw      = (float*)alloc((size_t)MM * NA * 4);
  float* entw     = (float*)alloc((size_t)MM * 4);
  float* actF     = (float*)alloc((size_t)MM * DD * 4);
  bf16*  actB     = (bf16*)alloc((size_t)MM * DD * 2);
  // Reuse dead regions: nodes_b+values_b (16MB contiguous) -> gate partial f32;
  // attn_b (8MB) -> silu hidden bf16. Both are dead by the time these are written.
  float* tmpF = (float*)nodes_b;
  bf16*  hB   = (bf16*)attn_b;

  dim3 blkMain(256), blkSmall(128), blkFlash(128);
  dim3 gMain(DD / 128, MM / 128);   // (8, 32)
  GemmEpi e;

  // 1-2) nodes / values projections (f32 A, bf16 out)
  e = GemmEpi{}; e.outB = nodes_b;
  gemm_wmma<EPI_BF16, false, 128, 128, 64, 32><<<gMain, blkMain, 0, stream>>>(x, W_node, DD, DD, e);
  e = GemmEpi{}; e.outB = values_b;
  gemm_wmma<EPI_BF16, false, 128, 128, 64, 32><<<gMain, blkMain, 0, stream>>>(x, W_value, DD, DD, e);

  // 3) ternary axis scores
  ax_kernel<<<dim3((BB * HH * TT) / 256), dim3(256), 0, stream>>>(nodes_b, arity_w, axw);

  // 4) fused attention (flash-style, never materializes scores)
  flash_attn<<<dim3(TT / 64, BB * HH), blkFlash, 0, stream>>>(nodes_b, values_b, axw, attn_b);

  // 5) x1 = x + attn_out @ W_out
  e = GemmEpi{}; e.outF = x1f; e.outB = x1b; e.addF = x;
  gemm_wmma<EPI_ADDX, true, 128, 128, 64, 32><<<gMain, blkMain, 0, stream>>>(attn_b, W_out, DD, DD, e);

  // 6) events = x1 @ W_event  (N=64 -> small-tile variant)
  e = GemmEpi{}; e.outF = events;
  gemm_wmma<EPI_F32, true, 64, 64, 32, 32><<<dim3(1, MM / 64), blkSmall, 0, stream>>>(x1b, W_event, DE, DD, e);

  // 7-8) rule bank: normalized patterns, top-4 soft hits, alt weights, entropy
  pn_prep<<<dim3(1), dim3(64), 0, stream>>>(patterns, pnw);
  rule_kernel<<<dim3(MM / 4), dim3(128), 0, stream>>>(events, pnw, patterns, W_alt, W_type,
                                                      x1b, log_temp, rsw, entw);

  // 9-10) actions_out = sum_a (alt_w[a]*sig(hit)) * (x1 @ W_actions[a])
  e = GemmEpi{}; e.outF = actF; e.rowscale = rsw + 0; e.rs_stride = NA;
  gemm_wmma<EPI_RS0, true, 128, 128, 64, 32><<<gMain, blkMain, 0, stream>>>(x1b, W_actions, DD, DD, e);
  e = GemmEpi{}; e.outF = actF; e.outB = actB; e.rowscale = rsw + 1; e.rs_stride = NA;
  gemm_wmma<EPI_RS1, true, 128, 128, 64, 32><<<gMain, blkMain, 0, stream>>>(x1b, W_actions + (size_t)DD * DD, DD, DD, e);

  // 11-12) gate hidden: silu(x1@Wg1[:D] + actions@Wg1[D:2D] + ent*Wg1[2D] + bg1)
  e = GemmEpi{}; e.outF = tmpF;
  gemm_wmma<EPI_F32, true, 128, 128, 64, 32><<<gMain, blkMain, 0, stream>>>(x1b, Wg1, DD, DD, e);
  e = GemmEpi{}; e.outB = hB; e.prevF = tmpF; e.ent = entw;
  e.wlast = Wg1 + (size_t)2 * DD * DD; e.bias = bg1;
  gemm_wmma<EPI_GATE, true, 128, 128, 64, 32><<<gMain, blkMain, 0, stream>>>(actB, Wg1 + (size_t)DD * DD, DD, DD, e);

  // 13) g = sigmoid(h@Wg2+bg2); out = x1 + g*actions_out
  final_kernel<<<dim3(MM / 8), dim3(256), 0, stream>>>(hB, Wg2, bg2, x1f, actF, out);
}
